// Voxel_Mamba_Waymo_4896262717900
// MI455X (gfx1250) — matrix operand order repaired and checked by reading, not compiled, not run on hardware
//
#include <hip/hip_runtime.h>
#include <hip/hip_bf16.h>
#include <stdint.h>

typedef _Float16 f16;
typedef __attribute__((ext_vector_type(16))) _Float16 v16h;
typedef __attribute__((ext_vector_type(8)))  _Float16 v8h;
typedef __attribute__((ext_vector_type(8)))  float    v8f;
typedef __attribute__((ext_vector_type(4)))  uint32_t v4u;
typedef __attribute__((ext_vector_type(8)))  uint32_t v8u;

#define NVOX 32768
#define DMODEL 128
#define DINNER 256
#define DSTATE 16
#define DTRANK 8
#define KCONV 4
#define NCHUNK 256
#define LCHUNK 128   // NVOX / NCHUNK
#define EPSV 1e-5f

// ---------------------------------------------------------------------------
// helpers
// ---------------------------------------------------------------------------
__device__ __forceinline__ float wave_sum32(float v) {
  #pragma unroll
  for (int off = 16; off > 0; off >>= 1) v += __shfl_xor(v, off, 32);
  return v;
}

__device__ __forceinline__ float fsilu(float x) {
  return x / (1.f + __expf(-x));
}

__device__ __forceinline__ float fsoftplus(float x) {
  return (x > 20.f) ? x : log1pf(__expf(x));
}

// WMMA A fragment: 16x32 f16, row m = lane&15.
// lanes 0-15 hold K 0..7 and 16..23 ; lanes 16-31 hold K 8..15 and 24..31.
__device__ __forceinline__ v16h load_a_frag(const f16* __restrict__ A, int lda,
                                            int m0, int k0, int lane) {
  const int m  = m0 + (lane & 15);
  const int kb = (lane >> 4) * 8;
  const f16* p = A + (size_t)m * lda + k0 + kb;
  v8h lo = *(const v8h*)(p);
  v8h hi = *(const v8h*)(p + 16);
  v16h r;
  #pragma unroll
  for (int i = 0; i < 8; ++i) { r[i] = lo[i]; r[8 + i] = hi[i]; }
  return r;
}

// WMMA B fragment: 32x16 f16, col n = lane&15.
// B is pre-transposed (Bt[n*ldb + k]) so each lane reads two runs of 8 contiguous halves (k+0..7, k+16..23).
// lanes 0-15 hold K 0..7 and 16..23 ; lanes 16-31 hold K 8..15 and 24..31.
__device__ __forceinline__ v16h load_b_frag(const f16* __restrict__ Bt, int ldb,
                                            int n0, int k0, int lane) {
  const int n  = n0 + (lane & 15);
  const int kb = (lane >> 4) * 8;
  const f16* p = Bt + (size_t)n * ldb + k0 + kb;
  v8h lo = *(const v8h*)(p);
  v8h hi = *(const v8h*)(p + 16);
  v16h r;
  #pragma unroll
  for (int i = 0; i < 8; ++i) { r[i] = lo[i]; r[8 + i] = hi[i]; }
  return r;
}

// ---------------------------------------------------------------------------
// Tensor Data Mover: 2D f32 tile (tile_w x tile_h elems, row stride
// stride_elems) from global -> contiguous LDS at lds_off.
// D# bitfields per CDNA5 ISA ch.8 (group0 128b, group1 256b); 2-group form
// (VADDR2/VADDR3 = NULL, <=2D tensor).
// ---------------------------------------------------------------------------
__device__ __forceinline__ void tdm_load_2d_f32(uint32_t lds_off, const void* gaddr,
                                                uint32_t tile_w, uint32_t tile_h,
                                                uint32_t stride_elems) {
  const uint64_t ga = (uint64_t)gaddr;
  v4u g0;
  g0[0] = 1u;                                        // count=1, user descriptor
  g0[1] = lds_off;                                   // lds_addr (bytes)
  g0[2] = (uint32_t)ga;                              // global_addr[31:0]
  g0[3] = (uint32_t)((ga >> 32) & 0x1FFFFFFu)        // global_addr[56:32]
          | (2u << 30);                              // type = 2 ("image")
  v8u g1;
  g1[0] = (2u << 16);                                // data_size = 2 (4 bytes)
  g1[1] = (tile_w & 0xFFFFu) << 16;                  // tensor_dim0[15:0]
  g1[2] = (tile_w >> 16) | ((tile_h & 0xFFFFu) << 16); // td0[31:16] | td1[15:0]
  g1[3] = (tile_h >> 16) | ((tile_w & 0xFFFFu) << 16); // td1[31:16] | tile_dim0
  g1[4] = tile_h & 0xFFFFu;                          // tile_dim1 (tile_dim2 = 0)
  g1[5] = stride_elems;                              // tensor_dim0_stride[31:0]
  g1[6] = 0u;                                        // stride0[47:32], stride1[15:0]
  g1[7] = 0u;                                        // stride1[47:16]
  asm volatile("tensor_load_to_lds %0, %1" :: "s"(g0), "s"(g1) : "memory");
}

// ---------------------------------------------------------------------------
// weight transpose + f32->f16 convert: dst[n*K + k] = src[k*Nc + n] (pad w/ 0)
// ---------------------------------------------------------------------------
__global__ void wcvt_kernel(const float* __restrict__ src, f16* __restrict__ dst,
                            int K, int Nc, int Npad) {
  int idx = blockIdx.x * blockDim.x + threadIdx.x;
  if (idx >= Npad * K) return;
  int n = idx / K;
  int k = idx - n * K;
  dst[(size_t)n * K + k] = (n < Nc) ? (f16)src[(size_t)k * Nc + n] : (f16)0.f;
}

// ---------------------------------------------------------------------------
// pos-encode + gather(perm) + residual + RMSNorm -> f16 activations
// one wave of 32 lanes per row (4 features / lane)
// ---------------------------------------------------------------------------
__global__ void prep_kernel(const float* __restrict__ vf,
                            const int*   __restrict__ coords,
                            const int*   __restrict__ perm,
                            const float* __restrict__ pos_w,
                            const float* __restrict__ pos_b,
                            const float* __restrict__ rms_w,
                            float* __restrict__ residual,
                            f16*   __restrict__ rmsh) {
  const int row  = blockIdx.x * 8 + (threadIdx.x >> 5);
  const int lane = threadIdx.x & 31;
  const int src  = perm[row];
  const int* c4  = coords + (size_t)src * 4;
  const int zi = c4[1], yi = c4[2], xi = c4[3];
  const float inv_denom = 1.f / 43.f;   // Yc//12 + 1 = 43
  float p[9];
  p[0] = (float)zi * (1.f / 16.f);
  p[1] = (float)(yi / 12) * inv_denom;
  p[2] = (float)(xi / 12) * inv_denom;
  p[3] = (float)(yi % 12) * (1.f / 12.f);
  p[4] = (float)(xi % 12) * (1.f / 12.f);
  p[5] = (float)((yi + 6) / 12) * inv_denom;
  p[6] = (float)((xi + 6) / 12) * inv_denom;
  p[7] = (float)((yi + 6) % 12) * (1.f / 12.f);
  p[8] = (float)((xi + 6) % 12) * (1.f / 12.f);

  float v[4];
  float ss = 0.f;
  #pragma unroll
  for (int q = 0; q < 4; ++q) {
    const int j = lane * 4 + q;
    float pe = pos_b[j];
    #pragma unroll
    for (int f = 0; f < 9; ++f) pe += p[f] * pos_w[f * DMODEL + j];
    const float val = vf[(size_t)src * DMODEL + j] + pe;
    v[q] = val;
    residual[(size_t)row * DMODEL + j] = val;
    ss += val * val;
  }
  ss = wave_sum32(ss);
  const float rn = rsqrtf(ss * (1.f / DMODEL) + EPSV);
  #pragma unroll
  for (int q = 0; q < 4; ++q) {
    const int j = lane * 4 + q;
    rmsh[(size_t)row * DMODEL + j] = (f16)(v[q] * rn * rms_w[j]);
  }
}

// ---------------------------------------------------------------------------
// f16 WMMA GEMM with N-blocking: each wave computes a 16 x (16*NBLK) strip.
// A fragment loaded once per K-step and reused for NBLK WMMAs.
// ---------------------------------------------------------------------------
template <int NBLK>
__global__ void gemm_wmma_kernel(const f16* __restrict__ A,
                                 const f16* __restrict__ Bt,
                                 float* __restrict__ Cout,
                                 const float* __restrict__ addsrc,
                                 int M, int Ntiles, int Kd, int ldc, int validN) {
  const int wave = (blockIdx.x * blockDim.x + threadIdx.x) >> 5;
  const int lane = threadIdx.x & 31;
  const int groups = Ntiles / NBLK;
  const int totalWaves = (M >> 4) * groups;
  if (wave >= totalWaves) return;
  const int tm  = (wave / groups) << 4;
  const int tn0 = (wave % groups) * (NBLK * 16);

  v8f c[NBLK];
  #pragma unroll
  for (int j = 0; j < NBLK; ++j) c[j] = (v8f){};

  for (int k0 = 0; k0 < Kd; k0 += 32) {
    if (k0 + 32 < Kd) {
      __builtin_prefetch(A + (size_t)(tm + (lane & 15)) * Kd + k0 + 32, 0, 3);
    }
    const v16h a = load_a_frag(A, Kd, tm, k0, lane);
    #pragma unroll
    for (int j = 0; j < NBLK; ++j) {
      const v16h b = load_b_frag(Bt, Kd, tn0 + j * 16, k0, lane);
      c[j] = __builtin_amdgcn_wmma_f32_16x16x32_f16(false, a, false, b,
                                                    (short)0, c[j], false, false);
    }
  }

  const int rbase = tm + ((lane >> 4) << 3);
  #pragma unroll
  for (int j = 0; j < NBLK; ++j) {
    const int col = tn0 + j * 16 + (lane & 15);
    if (col < validN) {
      #pragma unroll
      for (int r = 0; r < 8; ++r) {
        const int row = rbase + r;
        float v = c[j][r];
        if (addsrc) v += addsrc[(size_t)row * ldc + col];
        Cout[(size_t)row * ldc + col] = v;
      }
    }
  }
}

// ---------------------------------------------------------------------------
// causal depthwise conv (K=4) + SiLU ; reads x-half of xz (cols 0..255)
// ---------------------------------------------------------------------------
__global__ void conv_silu_kernel(const float* __restrict__ xz,
                                 const float* __restrict__ conv_w,
                                 const float* __restrict__ conv_b,
                                 float* __restrict__ xc,
                                 f16*   __restrict__ xch) {
  const int idx = blockIdx.x * blockDim.x + threadIdx.x;  // N * DINNER
  const int i = idx >> 8;
  const int d = idx & (DINNER - 1);
  float acc = conv_b[d];
  #pragma unroll
  for (int k = 0; k < KCONV; ++k) {
    const int t = i - (KCONV - 1) + k;
    if (t >= 0) acc += conv_w[d * KCONV + k] * xz[(size_t)t * (2 * DINNER) + d];
  }
  const float s = fsilu(acc);
  xc[idx]  = s;
  xch[idx] = (f16)s;
}

// ---------------------------------------------------------------------------
// dt = softplus(dbc[:, :8] @ dt_w + dt_b)
// ---------------------------------------------------------------------------
__global__ void dt_kernel(const float* __restrict__ dbc,
                          const float* __restrict__ dt_w,
                          const float* __restrict__ dt_b,
                          float* __restrict__ dt) {
  const int idx = blockIdx.x * blockDim.x + threadIdx.x;  // N * DINNER
  const int i = idx >> 8;
  const int d = idx & (DINNER - 1);
  float acc = dt_b[d];
  #pragma unroll
  for (int r = 0; r < DTRANK; ++r)
    acc += dbc[(size_t)i * 40 + r] * dt_w[r * DINNER + d];
  dt[idx] = fsoftplus(acc);
}

// ---------------------------------------------------------------------------
// scan phase A: per chunk c and channel pair (d,s): local decay product P and
// zero-init end state H over LCHUNK steps.  grid = (16, NCHUNK), block = 256
// ---------------------------------------------------------------------------
__global__ void scan_phaseA_kernel(const float* __restrict__ dt,
                                   const float* __restrict__ xc,
                                   const float* __restrict__ dbc,
                                   const float* __restrict__ A_log,
                                   float* __restrict__ chP,
                                   float* __restrict__ chH) {
  const int pair = blockIdx.x * blockDim.x + threadIdx.x;  // 0..4095
  const int c = blockIdx.y;
  const int d = pair >> 4;
  const int s = pair & (DSTATE - 1);
  const float Ads = -__expf(A_log[d * DSTATE + s]);
  float P = 1.f, H = 0.f;
  for (int t = 0; t < LCHUNK; ++t) {
    const int g = c * LCHUNK + t;
    const float dv = dt[(size_t)g * DINNER + d];
    const float xv = xc[(size_t)g * DINNER + d];
    const float bv = dbc[(size_t)g * 40 + DTRANK + s];
    const float a  = __expf(dv * Ads);
    P *= a;
    H = H * a + dv * xv * bv;
  }
  chP[(size_t)c * 4096 + pair] = P;
  chH[(size_t)c * 4096 + pair] = H;
}

// ---------------------------------------------------------------------------
// scan phase B: serial combine across chunks (depth NCHUNK). one block.
// ---------------------------------------------------------------------------
__global__ void scan_phaseB_kernel(const float* __restrict__ chP,
                                   const float* __restrict__ chH,
                                   float* __restrict__ hinit) {
  #pragma unroll
  for (int q = 0; q < 4; ++q) {
    const int pair = threadIdx.x + q * 1024;
    float h = 0.f;
    for (int c = 0; c < NCHUNK; ++c) {
      const size_t o = (size_t)c * 4096 + pair;
      hinit[o] = h;
      h = h * chP[o] + chH[o];
    }
  }
}

// ---------------------------------------------------------------------------
// scan phase C: replay chunk with correct init state; fuse y = ys + D*xc,
// y *= silu(z), emit f16 for out_proj.  grid = NCHUNK blocks, block = 256 (=d)
// B/C panel (128 rows x 32 floats, row-stride 40) staged to LDS via the
// Tensor Data Mover; wave 0 issues the DMA, waits TENSORcnt, then barrier.
// ---------------------------------------------------------------------------
__global__ void scan_phaseC_kernel(const float* __restrict__ dt,
                                   const float* __restrict__ xc,
                                   const float* __restrict__ dbc,
                                   const float* __restrict__ A_log,
                                   const float* __restrict__ Dskip,
                                   const float* __restrict__ xz,
                                   const float* __restrict__ hinit,
                                   f16* __restrict__ yh) {
  __shared__ float lBC[LCHUNK * 32];   // [t][0..15]=B, [t][16..31]=C
  const int c = blockIdx.x;
  const int d = threadIdx.x;

  if (threadIdx.x < 32) {
    const float* gsrc = dbc + (size_t)(c * LCHUNK) * 40 + DTRANK;
    tdm_load_2d_f32((uint32_t)(uintptr_t)&lBC[0], gsrc,
                    /*tile_w=*/32, /*tile_h=*/LCHUNK, /*stride_elems=*/40);
    __builtin_amdgcn_s_wait_tensorcnt(0);
  }
  __syncthreads();

  float h[DSTATE], Ads[DSTATE];
  #pragma unroll
  for (int s = 0; s < DSTATE; ++s) {
    h[s]   = hinit[(size_t)c * 4096 + d * DSTATE + s];
    Ads[s] = -__expf(A_log[d * DSTATE + s]);
  }
  const float dsk = Dskip[d];

  for (int t = 0; t < LCHUNK; ++t) {
    const int g = c * LCHUNK + t;
    const float dv = dt[(size_t)g * DINNER + d];
    const float xv = xc[(size_t)g * DINNER + d];
    const float dx = dv * xv;
    float acc = 0.f;
    #pragma unroll
    for (int s = 0; s < DSTATE; ++s) {
      const float a = __expf(dv * Ads[s]);
      h[s] = h[s] * a + dx * lBC[t * 32 + s];
      acc += h[s] * lBC[t * 32 + 16 + s];
    }
    float y = acc + dsk * xv;
    const float zv = xz[(size_t)g * (2 * DINNER) + DINNER + d];
    y *= fsilu(zv);
    yh[(size_t)g * DINNER + d] = (f16)y;
  }
}

// ---------------------------------------------------------------------------
// inv_perm gather + LayerNorm -> d_out. one wave per output row.
// ---------------------------------------------------------------------------
__global__ void final_ln_kernel(const float* __restrict__ preout,
                                const int*   __restrict__ inv_perm,
                                const float* __restrict__ ln_w,
                                const float* __restrict__ ln_b,
                                float* __restrict__ out) {
  const int row  = blockIdx.x * 8 + (threadIdx.x >> 5);
  const int lane = threadIdx.x & 31;
  const int src  = inv_perm[row];
  float v[4];
  float s = 0.f;
  #pragma unroll
  for (int q = 0; q < 4; ++q) {
    v[q] = preout[(size_t)src * DMODEL + lane * 4 + q];
    s += v[q];
  }
  s = wave_sum32(s);
  const float mu = s * (1.f / DMODEL);
  float s2 = 0.f;
  #pragma unroll
  for (int q = 0; q < 4; ++q) { const float dlt = v[q] - mu; s2 += dlt * dlt; }
  s2 = wave_sum32(s2);
  const float inv = rsqrtf(s2 * (1.f / DMODEL) + EPSV);
  #pragma unroll
  for (int q = 0; q < 4; ++q) {
    const int j = lane * 4 + q;
    out[(size_t)row * DMODEL + j] = (v[q] - mu) * inv * ln_w[j] + ln_b[j];
  }
}

// ---------------------------------------------------------------------------
// host launcher
// ---------------------------------------------------------------------------
extern "C" void kernel_launch(void* const* d_in, const int* in_sizes, int n_in,
                              void* d_out, int out_size, void* d_ws, size_t ws_size,
                              hipStream_t stream) {
  // setup_inputs() dict order
  const float* vf        = (const float*)d_in[0];   // voxel_features  N*128
  const int*   coords    = (const int*)  d_in[1];   // voxel_coords    N*4
  const int*   perm      = (const int*)  d_in[2];   // perm            N
  const int*   inv_perm  = (const int*)  d_in[3];   // inv_perm        N
  const float* pos_w     = (const float*)d_in[4];   // 9*128
  const float* pos_b     = (const float*)d_in[5];   // 128
  const float* rms_w     = (const float*)d_in[6];   // 128
  const float* in_proj_w = (const float*)d_in[7];   // 128*512
  const float* conv_w    = (const float*)d_in[8];   // 256*4
  const float* conv_b    = (const float*)d_in[9];   // 256
  const float* x_proj_w  = (const float*)d_in[10];  // 256*40
  const float* dt_w      = (const float*)d_in[11];  // 8*256
  const float* dt_b      = (const float*)d_in[12];  // 256
  const float* A_log     = (const float*)d_in[13];  // 256*16
  const float* Dskip     = (const float*)d_in[14];  // 256
  const float* out_proj_w= (const float*)d_in[15];  // 256*128
  const float* ln_w      = (const float*)d_in[16];  // 128
  const float* ln_b      = (const float*)d_in[17];  // 128
  float* out = (float*)d_out;

  // workspace carve-up (256B aligned)
  char* ws = (char*)d_ws;
  size_t off = 0;
  auto alloc = [&](size_t bytes) {
    char* p = ws + off;
    off += (bytes + 255) & ~(size_t)255;
    return p;
  };
  float* residual = (float*)alloc((size_t)NVOX * DMODEL * 4);
  f16*   rmsh     = (f16*)  alloc((size_t)NVOX * DMODEL * 2);
  f16*   inWt     = (f16*)  alloc((size_t)512 * 128 * 2);
  f16*   xprojWt  = (f16*)  alloc((size_t)48 * 256 * 2);
  f16*   outWt    = (f16*)  alloc((size_t)128 * 256 * 2);
  float* xz       = (float*)alloc((size_t)NVOX * 512 * 4);
  float* xc       = (float*)alloc((size_t)NVOX * DINNER * 4);
  f16*   xch      = (f16*)  alloc((size_t)NVOX * DINNER * 2);
  float* dbc      = (float*)alloc((size_t)NVOX * 40 * 4);
  float* dt       = (float*)alloc((size_t)NVOX * DINNER * 4);
  float* chP      = (float*)alloc((size_t)NCHUNK * 4096 * 4);
  float* chH      = (float*)alloc((size_t)NCHUNK * 4096 * 4);
  float* hinit    = (float*)alloc((size_t)NCHUNK * 4096 * 4);
  f16*   yh       = (f16*)  alloc((size_t)NVOX * DINNER * 2);
  float* preout   = (float*)alloc((size_t)NVOX * DMODEL * 4);
  (void)ws_size; (void)n_in; (void)in_sizes; (void)out_size;

  // 1) weight convert + transpose to f16
  wcvt_kernel<<<(512 * 128 + 255) / 256, 256, 0, stream>>>(in_proj_w, inWt, 128, 512, 512);
  wcvt_kernel<<<(48 * 256 + 255) / 256, 256, 0, stream>>>(x_proj_w, xprojWt, 256, 40, 48);
  wcvt_kernel<<<(128 * 256 + 255) / 256, 256, 0, stream>>>(out_proj_w, outWt, 256, 128, 128);

  // 2) pos-encode + gather + residual + RMSNorm
  prep_kernel<<<NVOX / 8, 256, 0, stream>>>(vf, coords, perm, pos_w, pos_b, rms_w,
                                            residual, rmsh);

  // 3) in_proj: xz = rmsh (N x 128) @ W (128 x 512)   [NBLK=4 -> 8 wave-groups]
  {
    const int waves = (NVOX / 16) * (32 / 4);
    gemm_wmma_kernel<4><<<(waves * 32 + 255) / 256, 256, 0, stream>>>(
        rmsh, inWt, xz, nullptr, NVOX, 32, DMODEL, 512, 512);
  }

  // 4) causal conv + SiLU
  conv_silu_kernel<<<(NVOX * DINNER) / 256, 256, 0, stream>>>(xz, conv_w, conv_b, xc, xch);

  // 5) x_proj: dbc = xc (N x 256) @ W (256 x 40), padded to 48 cols  [NBLK=3]
  {
    const int waves = (NVOX / 16) * (3 / 3);
    gemm_wmma_kernel<3><<<(waves * 32 + 255) / 256, 256, 0, stream>>>(
        xch, xprojWt, dbc, nullptr, NVOX, 3, DINNER, 40, 40);
  }

  // 6) dt = softplus(dbc[:, :8] @ dt_w + dt_b)
  dt_kernel<<<(NVOX * DINNER) / 256, 256, 0, stream>>>(dbc, dt_w, dt_b, dt);

  // 7) chunked parallel selective scan
  scan_phaseA_kernel<<<dim3(16, NCHUNK), 256, 0, stream>>>(dt, xc, dbc, A_log, chP, chH);
  scan_phaseB_kernel<<<1, 1024, 0, stream>>>(chP, chH, hinit);
  scan_phaseC_kernel<<<NCHUNK, 256, 0, stream>>>(dt, xc, dbc, A_log, Dskip, xz, hinit, yh);

  // 8) out_proj + residual: preout = yh (N x 256) @ W (256 x 128) + residual
  {
    const int waves = (NVOX / 16) * (8 / 4);
    gemm_wmma_kernel<4><<<(waves * 32 + 255) / 256, 256, 0, stream>>>(
        yh, outWt, preout, residual, NVOX, 8, DINNER, DMODEL, DMODEL);
  }

  // 9) inv_perm gather + LayerNorm
  final_ln_kernel<<<NVOX / 8, 256, 0, stream>>>(preout, inv_perm, ln_w, ln_b, out);
}
